// CodeFormer_31945966748181
// MI455X (gfx1250) — compile-verified, matrix operand order checked by reading, not run on hardware
//
#include <hip/hip_runtime.h>
#include <hip/hip_bf16.h>
#include <vector>
#include <math.h>

// ---------------- CDNA5 WMMA types ----------------
typedef __attribute__((ext_vector_type(16))) __bf16 v16bf;
typedef __attribute__((ext_vector_type(8)))  float  v8f;
typedef __attribute__((ext_vector_type(4)))  unsigned int u32x4;
typedef __attribute__((ext_vector_type(4)))  int i32x4;
typedef __attribute__((ext_vector_type(8)))  int i32x8;

#if __has_builtin(__builtin_amdgcn_global_load_async_to_lds_b32) && \
    __has_builtin(__builtin_amdgcn_s_wait_asynccnt)
#define HAVE_ASYNC 1
#else
#define HAVE_ASYNC 0
#endif

#if __has_builtin(__builtin_amdgcn_tensor_load_to_lds) && \
    __has_builtin(__builtin_amdgcn_s_wait_tensorcnt)
#define HAVE_TDM 1
#else
#define HAVE_TDM 0
#endif

typedef __attribute__((address_space(1))) int gas_i32;
typedef __attribute__((address_space(3))) int las_i32;

#if HAVE_ASYNC
static __device__ __forceinline__ void async_cp_b32(const float* g, float* l) {
  __builtin_amdgcn_global_load_async_to_lds_b32((gas_i32*)g, (las_i32*)l, 0, 0);
}
#endif

static __device__ __forceinline__ unsigned short f2bf(float f) {
  unsigned u = __builtin_bit_cast(unsigned, f);
  unsigned r = (u + 0x7FFFu + ((u >> 16) & 1u)) >> 16;
  return (unsigned short)r;
}
static __device__ __forceinline__ unsigned pack_bf2(float a, float b) {
  return (unsigned)f2bf(a) | ((unsigned)f2bf(b) << 16);
}

static __device__ __forceinline__ float act_apply(float v, int act) {
  if (act == 1) v = v / (1.f + __expf(-v));              // SiLU
  else if (act == 2) v = v > 0.f ? v : 0.2f * v;         // LeakyReLU(0.2)
  else if (act == 3) v = 0.5f * v * (1.f + erff(v * 0.70710678118654752f)); // exact GELU
  return v;
}

union Frag { unsigned u[8]; v16bf v; };

// ---------------- implicit-GEMM conv (NCHW, batch 1) ----------------
// Block tile 128(M=pixels) x 64(N=CO); 8 waves (4x2); wave tile 32x32 = 2x2 WMMA frags.
// A: im2col + fp32->bf16 scalar staging (needs zero-fill). B: TDM tensor_load_to_lds of the
// 64x32 fp32 weight sub-tile (row stride K, LDS pad 4dw/32dw) when in-bounds, scalar otherwise.
__global__ __launch_bounds__(256)
void conv_wmma_kernel(const float* __restrict__ X, const float* __restrict__ Wt,
                      const float* __restrict__ bias, const float* __restrict__ addsrc,
                      float* __restrict__ Y,
                      int CI, int CO, int IH, int IW, int OH, int OW,
                      int stride, int pad, int ks, int act) {
  const int M = OH * OW;
  const int K = CI * ks * ks;
  const int pBase = blockIdx.x * 128;
  const int nBase = blockIdx.y * 64;

  __shared__ unsigned short As[128][36];
  __shared__ float Bw[64][36];          // [co][k] fp32, 4-dword row pad (TDM pad_enable layout)

  const int tid  = threadIdx.x;
  const int lane = tid & 31;
  const int wv   = tid >> 5;        // 8 waves
  const int wm   = wv >> 1;         // 0..3 (M)
  const int wn   = wv & 1;          // 0..1 (N)
  const int hf   = lane >> 4;
  const int lr   = lane & 15;
  const int m0   = wm * 32;
  const int n0   = wn * 32;

  v8f acc[2][2] = {};
  const int ks2 = ks * ks;

  for (int kb = 0; kb < K; kb += 32) {
    // ---- A tile: 128 pixels x 32 K, im2col + zero-fill, fp32 -> bf16 ----
    for (int i = 0; i < 16; ++i) {
      int e  = tid + i * 256;
      int mm = e >> 5, kk = e & 31;
      int p  = pBase + mm;
      int k  = kb + kk;
      float v = 0.f;
      if (p < M && k < K) {
        int oh = p / OW, ow = p - oh * OW;
        int ci = k / ks2;
        int r  = k - ci * ks2;
        int kh = r / ks, kw = r - kh * ks;
        int ih = oh * stride + kh - pad;
        int iw = ow * stride + kw - pad;
        if (ih >= 0 && ih < IH && iw >= 0 && iw < IW)
          v = X[((size_t)ci * IH + ih) * IW + iw];
      }
      As[mm][kk] = f2bf(v);
    }
    // ---- B tile: 64 CO x 32 K fp32 sub-tile of [CO][K] ----
    const bool fullB = (nBase + 64 <= CO) && (kb + 32 <= K);
    (void)fullB;
#if HAVE_TDM
    if (fullB) {
      if (wv == 0) {
        unsigned lds_addr = (unsigned)(size_t)(las_i32*)&Bw[0][0];
        unsigned long long ga = (unsigned long long)(size_t)&Wt[(size_t)nBase * K + kb];
        u32x4 g0;
        g0[0] = 1u;                                            // count=1 (valid descriptor)
        g0[1] = lds_addr;                                      // lds_addr
        g0[2] = (unsigned)(ga & 0xFFFFFFFFu);                  // global_addr[31:0]
        g0[3] = (unsigned)((ga >> 32) & 0x1FFFFFFu) | (2u << 30); // global_addr[56:32], type=2
        i32x8 g1;
        g1[0] = (2 << 16) | (1 << 20) | (4 << 22) | (3 << 25); // 4B data, pad 4dw every 32dw
        unsigned td0 = 32u, td1 = 64u;                         // tensor dims (in-bounds tile)
        g1[1] = (int)((td0 & 0xFFFFu) << 16);                  // tensor_dim0[15:0]
        g1[2] = (int)((td0 >> 16) | ((td1 & 0xFFFFu) << 16));  // dim0[31:16], dim1[15:0]
        g1[3] = (int)((td1 >> 16) | (32u << 16));              // dim1[31:16], tile_dim0=32
        g1[4] = (int)64;                                       // tile_dim1=64, tile_dim2=0
        unsigned long long s0 = (unsigned long long)K;         // dim0 stride (elements)
        g1[5] = (int)(s0 & 0xFFFFFFFFu);
        g1[6] = (int)((s0 >> 32) & 0xFFFFu);                   // s0[47:32], dim1_stride lo16=0
        g1[7] = 0;
        i32x4 gz = {};
        i32x8 gz8 = {};
        __builtin_amdgcn_tensor_load_to_lds(g0, g1, gz, gz, gz8, 0);
      }
      __builtin_amdgcn_s_wait_tensorcnt(0);
    } else
#endif
    {
      for (int i = 0; i < 8; ++i) {
        int e  = tid + i * 256;
        int nn = e >> 5, kk = e & 31;
        int k  = kb + kk, n = nBase + nn;
        Bw[nn][kk] = (k < K && n < CO) ? Wt[(size_t)n * K + k] : 0.f;
      }
    }
    __syncthreads();

    Frag fa[2], fb[2];
#pragma unroll
    for (int v8 = 0; v8 < 8; ++v8) {
      int kk = ((v8 >> 2) * 16) + hf * 8 + ((v8 & 3) * 2);
      fa[0].u[v8] = *(const unsigned*)&As[m0 + lr][kk];
      fa[1].u[v8] = *(const unsigned*)&As[m0 + 16 + lr][kk];
      fb[0].u[v8] = pack_bf2(Bw[n0 + lr][kk],      Bw[n0 + lr][kk + 1]);
      fb[1].u[v8] = pack_bf2(Bw[n0 + 16 + lr][kk], Bw[n0 + 16 + lr][kk + 1]);
    }
    acc[0][0] = __builtin_amdgcn_wmma_f32_16x16x32_bf16(false, fa[0].v, false, fb[0].v, (short)0, acc[0][0], false, false);
    acc[0][1] = __builtin_amdgcn_wmma_f32_16x16x32_bf16(false, fa[0].v, false, fb[1].v, (short)0, acc[0][1], false, false);
    acc[1][0] = __builtin_amdgcn_wmma_f32_16x16x32_bf16(false, fa[1].v, false, fb[0].v, (short)0, acc[1][0], false, false);
    acc[1][1] = __builtin_amdgcn_wmma_f32_16x16x32_bf16(false, fa[1].v, false, fb[1].v, (short)0, acc[1][1], false, false);
    __syncthreads();
  }

#pragma unroll
  for (int im = 0; im < 2; ++im)
#pragma unroll
    for (int jn = 0; jn < 2; ++jn)
#pragma unroll
      for (int j = 0; j < 8; ++j) {
        int p = pBase + m0 + im * 16 + hf * 8 + j;
        int n = nBase + n0 + jn * 16 + lr;
        if (p < M && n < CO) {
          float v = acc[im][jn][j] + (bias ? bias[n] : 0.f);
          v = act_apply(v, act);
          if (addsrc) v += addsrc[(size_t)n * M + p];
          Y[(size_t)n * M + p] = v;
        }
      }
}

// ---------------- general WMMA GEMM ----------------
// Y[m][n] = act( alpha * sum_k A[m][k]*B[k][n] + bias[n] ) + addsrc[m][n]
// a_trans: A stored [K][M]; b_trans: B stored [N][K].
// Double-buffered async-to-LDS pipeline when tile-aligned (all GEMMs in this net are).
__global__ __launch_bounds__(256)
void gemm_wmma_kernel(const float* __restrict__ A, const float* __restrict__ B,
                      const float* __restrict__ bias, const float* __restrict__ addsrc,
                      float* __restrict__ Y,
                      int M, int N, int K, int lda, int ldb, int ldc,
                      int a_trans, int b_trans, float alpha, int act) {
  const int pBase = blockIdx.x * 128;
  const int nBase = blockIdx.y * 64;

  __shared__ float As[2][128][36];
  __shared__ float Bs[2][32][68];

  const int tid  = threadIdx.x;
  const int lane = tid & 31;
  const int wv   = tid >> 5;
  const int wm   = wv >> 1;
  const int wn   = wv & 1;
  const int hf   = lane >> 4;
  const int lr   = lane & 15;
  const int m0   = wm * 32;
  const int n0   = wn * 32;

  v8f acc[2][2] = {};

  auto compute_tile = [&](int buf) {
    Frag fa[2], fb[2];
#pragma unroll
    for (int v8 = 0; v8 < 8; ++v8) {
      int kk = ((v8 >> 2) * 16) + hf * 8 + ((v8 & 3) * 2);
      fa[0].u[v8] = pack_bf2(As[buf][m0 + lr][kk],      As[buf][m0 + lr][kk + 1]);
      fa[1].u[v8] = pack_bf2(As[buf][m0 + 16 + lr][kk], As[buf][m0 + 16 + lr][kk + 1]);
      fb[0].u[v8] = pack_bf2(Bs[buf][kk][n0 + lr],      Bs[buf][kk + 1][n0 + lr]);
      fb[1].u[v8] = pack_bf2(Bs[buf][kk][n0 + 16 + lr], Bs[buf][kk + 1][n0 + 16 + lr]);
    }
    acc[0][0] = __builtin_amdgcn_wmma_f32_16x16x32_bf16(false, fa[0].v, false, fb[0].v, (short)0, acc[0][0], false, false);
    acc[0][1] = __builtin_amdgcn_wmma_f32_16x16x32_bf16(false, fa[0].v, false, fb[1].v, (short)0, acc[0][1], false, false);
    acc[1][0] = __builtin_amdgcn_wmma_f32_16x16x32_bf16(false, fa[1].v, false, fb[0].v, (short)0, acc[1][0], false, false);
    acc[1][1] = __builtin_amdgcn_wmma_f32_16x16x32_bf16(false, fa[1].v, false, fb[1].v, (short)0, acc[1][1], false, false);
  };

#if HAVE_ASYNC
  const bool aligned = ((M & 127) == 0) && ((N & 63) == 0) && ((K & 31) == 0);
  auto load_async = [&](int kb, int buf) {
    for (int i = 0; i < 16; ++i) {
      int e  = tid + i * 256;
      int mm = e >> 5, kk = e & 31;
      const float* g = a_trans ? &A[(size_t)(kb + kk) * lda + (pBase + mm)]
                               : &A[(size_t)(pBase + mm) * lda + (kb + kk)];
      async_cp_b32(g, &As[buf][mm][kk]);
    }
    for (int i = 0; i < 8; ++i) {
      int e  = tid + i * 256;
      int kk = e >> 6, nn = e & 63;
      const float* g = b_trans ? &B[(size_t)(nBase + nn) * ldb + (kb + kk)]
                               : &B[(size_t)(kb + kk) * ldb + (nBase + nn)];
      async_cp_b32(g, &Bs[buf][kk][nn]);
    }
  };
  if (aligned) {
    // software pipeline: prefetch tile k+1 (24 async b32 per wave) while computing tile k
    load_async(0, 0);
    int curb = 0;
    for (int kb = 0; kb < K; kb += 32) {
      bool hn = (kb + 32) < K;
      if (hn) {
        load_async(kb + 32, curb ^ 1);
        __builtin_amdgcn_s_wait_asynccnt(24);   // exactly next tile outstanding -> cur done
      } else {
        __builtin_amdgcn_s_wait_asynccnt(0);
      }
      __syncthreads();
      compute_tile(curb);
      __syncthreads();
      curb ^= 1;
    }
  } else
#endif
  {
    for (int kb = 0; kb < K; kb += 32) {
      for (int i = 0; i < 16; ++i) {
        int e  = tid + i * 256;
        int mm = e >> 5, kk = e & 31;
        int m  = pBase + mm, k = kb + kk;
        float v = 0.f;
        if (m < M && k < K)
          v = a_trans ? A[(size_t)k * lda + m] : A[(size_t)m * lda + k];
        As[0][mm][kk] = v;
      }
      for (int i = 0; i < 8; ++i) {
        int e  = tid + i * 256;
        int kk = e >> 6, nn = e & 63;
        int k  = kb + kk, n = nBase + nn;
        float v = 0.f;
        if (k < K && n < N)
          v = b_trans ? B[(size_t)n * ldb + k] : B[(size_t)k * ldb + n];
        Bs[0][kk][nn] = v;
      }
      __syncthreads();
      compute_tile(0);
      __syncthreads();
    }
  }

#pragma unroll
  for (int im = 0; im < 2; ++im)
#pragma unroll
    for (int jn = 0; jn < 2; ++jn)
#pragma unroll
      for (int j = 0; j < 8; ++j) {
        int m = pBase + m0 + im * 16 + hf * 8 + j;
        int n = nBase + n0 + jn * 16 + lr;
        if (m < M && n < N) {
          float v = alpha * acc[im][jn][j];
          if (bias) v += bias[n];
          v = act_apply(v, act);
          if (addsrc) v += addsrc[(size_t)m * ldc + n];
          Y[(size_t)m * ldc + n] = v;
        }
      }
}

// ---------------- normalization / softmax / misc ----------------
__global__ __launch_bounds__(256)
void gn_kernel(const float* __restrict__ x, const float* __restrict__ g,
               const float* __restrict__ b, float* __restrict__ y,
               int C, int HW, int do_silu) {
  int grp = blockIdx.x;
  int cpg = C / 32;
  size_t base = (size_t)grp * cpg * HW;
  int N = cpg * HW;
  __shared__ float r1[256], r2[256];
  float s = 0.f, s2 = 0.f;
  for (int i = threadIdx.x; i < N; i += 256) { float v = x[base + i]; s += v; s2 += v * v; }
  r1[threadIdx.x] = s; r2[threadIdx.x] = s2; __syncthreads();
  for (int o = 128; o > 0; o >>= 1) {
    if (threadIdx.x < o) { r1[threadIdx.x] += r1[threadIdx.x + o]; r2[threadIdx.x] += r2[threadIdx.x + o]; }
    __syncthreads();
  }
  float mean = r1[0] / N;
  float var  = r2[0] / N - mean * mean;
  float rs   = rsqrtf(var + 1e-6f);
  for (int i = threadIdx.x; i < N; i += 256) {
    int c = grp * cpg + i / HW;
    float v = (x[base + i] - mean) * rs * g[c] + b[c];
    if (do_silu) v = v / (1.f + __expf(-v));
    y[base + i] = v;
  }
}

__global__ __launch_bounds__(256)
void ln_kernel(const float* __restrict__ x, const float* __restrict__ g,
               const float* __restrict__ b, float* __restrict__ y, int E) {
  int row = blockIdx.x;
  const float* xr = x + (size_t)row * E;
  __shared__ float r1[256], r2[256];
  float s = 0.f, s2 = 0.f;
  for (int i = threadIdx.x; i < E; i += 256) { float v = xr[i]; s += v; s2 += v * v; }
  r1[threadIdx.x] = s; r2[threadIdx.x] = s2; __syncthreads();
  for (int o = 128; o > 0; o >>= 1) {
    if (threadIdx.x < o) { r1[threadIdx.x] += r1[threadIdx.x + o]; r2[threadIdx.x] += r2[threadIdx.x + o]; }
    __syncthreads();
  }
  float mean = r1[0] / E;
  float var  = r2[0] / E - mean * mean;
  float rs   = rsqrtf(var + 1e-5f);
  for (int i = threadIdx.x; i < E; i += 256)
    y[(size_t)row * E + i] = (xr[i] - mean) * rs * g[i] + b[i];
}

__global__ __launch_bounds__(256)
void softmax_kernel(float* __restrict__ x, int C) {
  int row = blockIdx.x;
  float* xr = x + (size_t)row * C;
  __shared__ float red[256];
  float m = -1e30f;
  for (int i = threadIdx.x; i < C; i += 256) m = fmaxf(m, xr[i]);
  red[threadIdx.x] = m; __syncthreads();
  for (int o = 128; o > 0; o >>= 1) {
    if (threadIdx.x < o) red[threadIdx.x] = fmaxf(red[threadIdx.x], red[threadIdx.x + o]);
    __syncthreads();
  }
  m = red[0]; __syncthreads();
  float s = 0.f;
  for (int i = threadIdx.x; i < C; i += 256) { float e = __expf(xr[i] - m); xr[i] = e; s += e; }
  red[threadIdx.x] = s; __syncthreads();
  for (int o = 128; o > 0; o >>= 1) {
    if (threadIdx.x < o) red[threadIdx.x] += red[threadIdx.x + o];
    __syncthreads();
  }
  float inv = 1.f / red[0];
  for (int i = threadIdx.x; i < C; i += 256) xr[i] *= inv;
}

__global__ __launch_bounds__(256)
void add_kernel(const float* __restrict__ a, const float* __restrict__ b,
                float* __restrict__ y, int n) {
  int t = blockIdx.x * 256 + threadIdx.x;
  if (t < n) y[t] = a[t] + b[t];
}

__global__ __launch_bounds__(256)
void upsample_kernel(const float* __restrict__ x, float* __restrict__ y,
                     int C, int H, int W) {
  int OW = 2 * W, OHt = 2 * H;
  int total = C * OHt * OW;
  int t = blockIdx.x * 256 + threadIdx.x;
  if (t < total) {
    int ow = t % OW;
    int tmp = t / OW;
    int oh = tmp % OHt;
    int c  = tmp / OHt;
    y[t] = x[((size_t)c * H + (oh >> 1)) * W + (ow >> 1)];
  }
}

__global__ __launch_bounds__(256)
void concat_kernel(const float* __restrict__ a, const float* __restrict__ b,
                   float* __restrict__ y, int C1, int C2, int HW) {
  int total = (C1 + C2) * HW;
  int t = blockIdx.x * 256 + threadIdx.x;
  if (t < total) {
    int c = t / HW, p = t - c * HW;
    y[t] = (c < C1) ? a[(size_t)c * HW + p] : b[(size_t)(c - C1) * HW + p];
  }
}

__global__ __launch_bounds__(256)
void argmax_kernel(const float* __restrict__ logits, int* __restrict__ idx, int C) {
  int row = blockIdx.x;
  const float* xr = logits + (size_t)row * C;
  __shared__ float bv[256]; __shared__ int bi[256];
  float best = -1e30f; int bidx = 0;
  for (int i = threadIdx.x; i < C; i += 256) {
    float v = xr[i];
    if (v > best) { best = v; bidx = i; }
  }
  bv[threadIdx.x] = best; bi[threadIdx.x] = bidx; __syncthreads();
  for (int o = 128; o > 0; o >>= 1) {
    if (threadIdx.x < o) {
      if (bv[threadIdx.x + o] > bv[threadIdx.x] ||
          (bv[threadIdx.x + o] == bv[threadIdx.x] && bi[threadIdx.x + o] < bi[threadIdx.x])) {
        bv[threadIdx.x] = bv[threadIdx.x + o];
        bi[threadIdx.x] = bi[threadIdx.x + o];
      }
    }
    __syncthreads();
  }
  if (threadIdx.x == 0) idx[row] = bi[0];
}

__global__ __launch_bounds__(256)
void gather_kernel(const float* __restrict__ codebook, const int* __restrict__ idx,
                   float* __restrict__ quant, int L, int E) {
  int t = blockIdx.x * 256 + threadIdx.x;
  if (t < L * E) {
    int c = t / L, p = t - c * L;
    quant[t] = codebook[(size_t)idx[p] * E + c];   // [E][L] == NCHW (1,E,16,16)
  }
}

// ---------------- host orchestration ----------------
static inline int cdiv(int a, int b) { return (a + b - 1) / b; }

struct ResP  { const float* p[10]; bool sw; };
struct AttnP { const float* p[10]; };
struct CP    { const float* w; const float* b; };
struct Blk   { int t; int ci, co; ResP r; AttnP a; CP c; }; // t:0 conv3x3,1 res,2 attn,3 down,4 up
struct FuseP { const float *w1, *b1, *w2, *b2; };
struct TxfP  { const float* p[16]; };

extern "C" void kernel_launch(void* const* d_in, const int* in_sizes, int n_in,
                              void* d_out, int out_size, void* d_ws, size_t ws_size,
                              hipStream_t stream) {
  (void)in_sizes; (void)n_in; (void)out_size; (void)ws_size;

  // ---- parameter walk: setup_inputs() construction order ----
  int cur = 0;
  auto P = [&]() { return (const float*)d_in[cur++]; };
  const float* x_in = P();

  std::vector<Blk> enc, dec;
  auto takeRes = [&](int ci, int co) {
    Blk b{}; b.t = 1; b.ci = ci; b.co = co;
    for (int i = 0; i < 8; ++i) b.r.p[i] = P();
    b.r.sw = (ci != co);
    if (b.r.sw) { b.r.p[8] = P(); b.r.p[9] = P(); }
    return b;
  };
  auto takeAttn = [&](int cch) {
    Blk b{}; b.t = 2; b.ci = b.co = cch;
    for (int i = 0; i < 10; ++i) b.a.p[i] = P();
    return b;
  };
  auto takeC = [&](int t, int ci, int co) {
    Blk b{}; b.t = t; b.ci = ci; b.co = co;
    b.c.w = P(); b.c.b = P();
    return b;
  };

  static const int CH[6] = {1, 2, 2, 4, 4, 8};
  static const int IM[7] = {1, 1, 2, 2, 4, 4, 8};
  enc.push_back(takeC(0, 3, 64));
  {
    int curr = 512;
    for (int i = 0; i < 6; ++i) {
      int ci = 64 * IM[i], co = 64 * CH[i];
      for (int r = 0; r < 2; ++r) {
        enc.push_back(takeRes(ci, co)); ci = co;
        if (curr == 16) enc.push_back(takeAttn(co));
      }
      if (i != 5) { enc.push_back(takeC(3, ci, ci)); curr /= 2; }
    }
  }
  enc.push_back(takeRes(512, 512));
  enc.push_back(takeAttn(512));
  enc.push_back(takeRes(512, 512));
  const float *et_ng = P(), *et_nb = P(), *et_w = P(), *et_b = P();

  dec.push_back(takeC(0, 256, 512));
  dec.push_back(takeRes(512, 512));
  dec.push_back(takeAttn(512));
  dec.push_back(takeRes(512, 512));
  {
    int bi = 512, curr = 16;
    for (int i = 5; i >= 0; --i) {
      int co = 64 * CH[i];
      for (int r = 0; r < 3; ++r) {
        dec.push_back(takeRes(bi, co)); bi = co;
        if (curr == 16) dec.push_back(takeAttn(co));
      }
      if (i != 0) { dec.push_back(takeC(4, co, co)); curr *= 2; }
    }
  }
  const float *dt_ng = P(), *dt_nb = P(), *dt_w = P(), *dt_b = P();

  FuseP fe[3], fg[3];                         // keys '32','64','128' in order
  for (int i = 0; i < 3; ++i) { fe[i].w1 = P(); fe[i].b1 = P(); fe[i].w2 = P(); fe[i].b2 = P(); }
  for (int i = 0; i < 3; ++i) { fg[i].w1 = P(); fg[i].b1 = P(); fg[i].w2 = P(); fg[i].b2 = P(); }
  const float* pos    = P();
  const float* feat_w = P();
  const float* feat_b = P();
  TxfP ly[9];
  for (int l = 0; l < 9; ++l)
    for (int i = 0; i < 16; ++i) ly[l].p[i] = P();
  const float *idx_ng = P(), *idx_nb = P(), *idx_w = P();
  const float* codebook = P();

  // ---- workspace layout ----
  char* wp = (char*)d_ws;
  size_t off = 0;
  auto WSA = [&](size_t bytes) -> float* {
    float* p = (float*)(wp + off);
    off = (off + bytes + 255) & ~(size_t)255;
    return p;
  };
  const size_t BIG = (size_t)128 * 512 * 512 * sizeof(float);
  float* P0 = WSA(BIG);
  float* P1 = WSA(BIG);
  float* P2 = WSA(BIG);
  float* P3 = WSA(BIG);
  float* CC = WSA((size_t)32 * 1024 * 1024);
  float* S128 = WSA((size_t)128 * 128 * 128 * sizeof(float));
  float* S64  = WSA((size_t)256 * 64 * 64 * sizeof(float));
  float* S32  = WSA((size_t)256 * 32 * 32 * sizeof(float));
  float* Tlq  = WSA(65536 * sizeof(float));
  float* Txq  = WSA(131072 * sizeof(float));
  float* Th5  = WSA(131072 * sizeof(float));
  float* Tqk  = WSA(131072 * sizeof(float));
  float* Tq2  = WSA(131072 * sizeof(float));
  float* Tk2  = WSA(131072 * sizeof(float));
  float* Tv2  = WSA(131072 * sizeof(float));
  float* To   = WSA(131072 * sizeof(float));
  float* Tx2  = WSA(131072 * sizeof(float));
  float* Tffn = WSA(262144 * sizeof(float));
  float* Tatt = WSA(65536 * sizeof(float));
  float* Tlog = WSA(262144 * sizeof(float));
  float* Tquant = WSA(65536 * sizeof(float));
  int*   Tidx = (int*)WSA(256 * sizeof(int));

  float* pool[3] = {P0, P1, P2};

  // ---- launch helpers ----
  auto conv = [&](const float* X, const float* Wc, const float* Bi, const float* add,
                  float* Y, int CI, int CO, int IH, int IW,
                  int stride, int pad, int ks, int act, int OH, int OW) {
    dim3 g((unsigned)cdiv(OH * OW, 128), (unsigned)cdiv(CO, 64));
    conv_wmma_kernel<<<g, 256, 0, stream>>>(X, Wc, Bi, add, Y, CI, CO, IH, IW, OH, OW,
                                            stride, pad, ks, act);
  };
  auto gemm = [&](const float* A, const float* B, const float* bias, const float* add,
                  float* Y, int M, int N, int K, int lda, int ldb, int ldc,
                  int at, int bt, float alpha, int act) {
    dim3 g((unsigned)cdiv(M, 128), (unsigned)cdiv(N, 64));
    gemm_wmma_kernel<<<g, 256, 0, stream>>>(A, B, bias, add, Y, M, N, K, lda, ldb, ldc,
                                            at, bt, alpha, act);
  };
  auto gn = [&](const float* x, const float* g, const float* b, float* y,
                int C, int HW, int silu) {
    gn_kernel<<<32, 256, 0, stream>>>(x, g, b, y, C, HW, silu);
  };
  auto ln = [&](const float* x, const float* g, const float* b, float* y, int R, int E) {
    ln_kernel<<<(unsigned)R, 256, 0, stream>>>(x, g, b, y, E);
  };
  auto softmax = [&](float* x, int R, int C) {
    softmax_kernel<<<(unsigned)R, 256, 0, stream>>>(x, C);
  };
  auto addk = [&](const float* a, const float* b, float* y, int n) {
    add_kernel<<<(unsigned)cdiv(n, 256), 256, 0, stream>>>(a, b, y, n);
  };

  // ---- activation state ----
  const float* h = x_in;
  int hslot = -1;
  int C = 3, H = 512, Wd = 512;
  auto other = [&](int a, int b) { for (int s = 0; s < 3; ++s) if (s != a && s != b) return s; return 0; };

  auto res_exec = [&](const ResP& r, int ci, int co) {
    int A  = other(hslot, -1);
    gn(h, r.p[0], r.p[1], pool[A], ci, H * Wd, 1);
    int B2 = other(hslot, A);
    conv(pool[A], r.p[2], r.p[3], nullptr, pool[B2], ci, co, H, Wd, 1, 1, 3, 0, H, Wd);
    gn(pool[B2], r.p[4], r.p[5], pool[A], co, H * Wd, 1);
    const float* scp = h;
    if (r.sw) { conv(h, r.p[8], r.p[9], nullptr, P3, ci, co, H, Wd, 1, 0, 1, 0, H, Wd); scp = P3; }
    conv(pool[A], r.p[6], r.p[7], scp, pool[B2], co, co, H, Wd, 1, 1, 3, 0, H, Wd);
    h = pool[B2]; hslot = B2; C = co;
  };

  auto attn_exec = [&](const AttnP& a) {
    float scale = 1.f / sqrtf((float)C);
    int A = other(hslot, -1);
    gn(h, a.p[0], a.p[1], pool[A], C, 256, 0);
    conv(pool[A], a.p[2], a.p[3], nullptr, Tq2, C, C, 16, 16, 1, 0, 1, 0, 16, 16);
    conv(pool[A], a.p[4], a.p[5], nullptr, Tk2, C, C, 16, 16, 1, 0, 1, 0, 16, 16);
    conv(pool[A], a.p[6], a.p[7], nullptr, Tv2, C, C, 16, 16, 1, 0, 1, 0, 16, 16);
    // att[i][j] = softmax_j( sum_c q[c][i] k[c][j] * C^-0.5 )
    gemm(Tq2, Tk2, nullptr, nullptr, Tatt, 256, 256, C, 256, 256, 256, 1, 0, scale, 0);
    softmax(Tatt, 256, 256);
    // hnew[c][i] = sum_j v[c][j] att[i][j]
    gemm(Tv2, Tatt, nullptr, nullptr, Th5, C, 256, 256, 256, 256, 256, 0, 1, 1.f, 0);
    int B2 = other(hslot, A);
    conv(Th5, a.p[8], a.p[9], h, pool[B2], C, C, 16, 16, 1, 0, 1, 0, 16, 16);
    h = pool[B2]; hslot = B2;
  };

  auto apply_blk = [&](const Blk& b) {
    if (b.t == 0) {                       // conv_in 3x3
      int B2 = other(hslot, -1);
      conv(h, b.c.w, b.c.b, nullptr, pool[B2], b.ci, b.co, H, Wd, 1, 1, 3, 0, H, Wd);
      h = pool[B2]; hslot = B2; C = b.co;
    } else if (b.t == 1) {
      res_exec(b.r, b.ci, b.co);
    } else if (b.t == 2) {
      attn_exec(b.a);
    } else if (b.t == 3) {                // downsample: pad(0,1,0,1), stride-2 valid
      int B2 = other(hslot, -1);
      conv(h, b.c.w, b.c.b, nullptr, pool[B2], C, C, H, Wd, 2, 0, 3, 0, H / 2, Wd / 2);
      h = pool[B2]; hslot = B2; H /= 2; Wd /= 2;
    } else {                              // upsample 2x nearest + conv3x3
      int A = other(hslot, -1);
      upsample_kernel<<<(unsigned)cdiv(C * 4 * H * Wd, 256), 256, 0, stream>>>(h, pool[A], C, H, Wd);
      H *= 2; Wd *= 2;
      int B2 = other(hslot, A);
      conv(pool[A], b.c.w, b.c.b, nullptr, pool[B2], C, C, H, Wd, 1, 1, 3, 0, H, Wd);
      h = pool[B2]; hslot = B2;
    }
  };

  auto fuse_enc_exec = [&](const FuseP& f, float* dst, int cch) {
    int A = other(hslot, -1);
    conv(h, f.w1, f.b1, nullptr, pool[A], cch, cch, H, Wd, 1, 1, 3, 2, H, Wd);
    conv(pool[A], f.w2, f.b2, nullptr, dst, cch, cch, H, Wd, 1, 1, 3, 0, H, Wd);
  };
  auto fuse_gen_exec = [&](const FuseP& f, const float* skip, int cch) {
    int HW = H * Wd;
    concat_kernel<<<(unsigned)cdiv(2 * cch * HW, 256), 256, 0, stream>>>(skip, h, CC, cch, cch, HW);
    int A = other(hslot, -1);
    conv(CC, f.w1, f.b1, nullptr, pool[A], 2 * cch, cch, H, Wd, 1, 1, 3, 2, H, Wd);
    int B2 = other(hslot, A);
    conv(pool[A], f.w2, f.b2, h, pool[B2], cch, cch, H, Wd, 1, 1, 3, 0, H, Wd);
    h = pool[B2]; hslot = B2;
  };

  // ================= encoder =================
  for (size_t ib = 0; ib < enc.size(); ++ib) {
    apply_blk(enc[ib]);
    if (ib == 8)       fuse_enc_exec(fe[2], S128, 128);   // '128'
    else if (ib == 11) fuse_enc_exec(fe[1], S64, 256);    // '64'
    else if (ib == 14) fuse_enc_exec(fe[0], S32, 256);    // '32'
  }
  {
    int A = other(hslot, -1);
    gn(h, et_ng, et_nb, pool[A], 512, 256, 1);
    conv(pool[A], et_w, et_b, nullptr, Tlq, 512, 256, 16, 16, 1, 1, 3, 0, 16, 16);  // lq [256][256]
  }

  // ================= transformer =================
  // tokens = lq^T  -> x = tok @ feat_w + feat_b
  gemm(Tlq, feat_w, feat_b, nullptr, Txq, 256, 512, 256, 256, 512, 512, 1, 0, 1.f, 0);
  for (int l = 0; l < 9; ++l) {
    const TxfP& p = ly[l];
    ln(Txq, p.p[0], p.p[1], Th5, 256, 512);
    addk(Th5, pos, Tqk, 256 * 512);
    gemm(Tqk, p.p[4], p.p[5], nullptr, Tq2, 256, 512, 512, 512, 512, 512, 0, 0, 1.f, 0);
    gemm(Tqk, p.p[6], p.p[7], nullptr, Tk2, 256, 512, 512, 512, 512, 512, 0, 0, 1.f, 0);
    gemm(Th5, p.p[8], p.p[9], nullptr, Tv2, 256, 512, 512, 512, 512, 512, 0, 0, 1.f, 0);
    for (int hd = 0; hd < 8; ++hd) {
      gemm(Tq2 + hd * 64, Tk2 + hd * 64, nullptr, nullptr, Tatt,
           256, 256, 64, 512, 512, 256, 0, 1, 0.125f, 0);
      softmax(Tatt, 256, 256);
      gemm(Tatt, Tv2 + hd * 64, nullptr, nullptr, To + hd * 64,
           256, 64, 256, 256, 512, 512, 0, 0, 1.f, 0);
    }
    gemm(To, p.p[10], p.p[11], Txq, Tx2, 256, 512, 512, 512, 512, 512, 0, 0, 1.f, 0);
    ln(Tx2, p.p[2], p.p[3], Th5, 256, 512);
    gemm(Th5, p.p[12], p.p[13], nullptr, Tffn, 256, 1024, 512, 512, 1024, 1024, 0, 0, 1.f, 3);
    gemm(Tffn, p.p[14], p.p[15], Tx2, Txq, 256, 512, 1024, 1024, 512, 512, 0, 0, 1.f, 0);
  }
  ln(Txq, idx_ng, idx_nb, Th5, 256, 512);
  gemm(Th5, idx_w, nullptr, nullptr, Tlog, 256, 1024, 512, 512, 1024, 1024, 0, 0, 1.f, 0);
  argmax_kernel<<<256, 256, 0, stream>>>(Tlog, Tidx, 1024);
  gather_kernel<<<(unsigned)cdiv(65536, 256), 256, 0, stream>>>(codebook, Tidx, Tquant, 256, 256);

  // ================= decoder =================
  h = Tquant; hslot = -1; C = 256; H = 16; Wd = 16;
  for (size_t ib = 0; ib < dec.size(); ++ib) {
    apply_blk(dec[ib]);
    if (ib == 11)      fuse_gen_exec(fg[0], S32, 256);    // '32'
    else if (ib == 15) fuse_gen_exec(fg[1], S64, 256);    // '64'
    else if (ib == 19) fuse_gen_exec(fg[2], S128, 128);   // '128'
  }
  {
    int A = other(hslot, -1);
    gn(h, dt_ng, dt_nb, pool[A], 64, H * Wd, 1);
    conv(pool[A], dt_w, dt_b, nullptr, (float*)d_out, 64, 3, 512, 512, 1, 1, 3, 0, 512, 512);
  }
}
